// ArcFace_41532333752747
// MI455X (gfx1250) — compile-verified, matrix operand order checked by reading, not run on hardware
//
#include <hip/hip_runtime.h>

typedef _Float16 f16;
typedef __attribute__((ext_vector_type(16))) _Float16 v16h;
typedef __attribute__((ext_vector_type(8)))  float    v8f;

#define N_ROWS 512
#define DDIM   512
#define NCLS   100000

#define BM 256
#define BN 128
#define BK 32
#define LDSS 40   // padded f16 row stride in LDS (80B: 16B-aligned, conflict-breaking)
#define TILE_ELEMS ((BM + BN) * LDSS)   // one double-buffer slice (A then B)

// margin constants (m = 0.5)
#define SCALE_F  64.0f
#define COSM     0.87758256189037271f   // cos(0.5)
#define SINM     0.47942553860420301f   // sin(0.5)
#define THRESH  (-0.87758256189037271f) // cos(pi - 0.5)
#define MMV      0.23971276930210151f   // sin(pi - 0.5) * 0.5

union Frag { v16h v; float4 q[2]; };

__device__ __forceinline__ float wave_reduce_sum(float v) {
#pragma unroll
  for (int o = 16; o > 0; o >>= 1) v += __shfl_down(v, o, 32);
  return v;
}

// LDS aperture: flat addr[31:0] is the DS byte offset.
__device__ __forceinline__ uint32_t lds_addr(const void* p) {
  return (uint32_t)(uintptr_t)p;
}

// CDNA5 async global->LDS copy (16B per lane), tracked by ASYNCcnt.
__device__ __forceinline__ void async_b128(uint32_t lds, const f16* g) {
  asm volatile("global_load_async_to_lds_b128 %0, %1, off"
               :: "v"(lds), "v"(g) : "memory");
}

__device__ __forceinline__ void wait_async0() {
  asm volatile("s_wait_asynccnt 0" ::: "memory");
}

// One block (256 threads) per row; rows [0,512) come from x, rest from weight.
// Writes L2-normalized rows as f16 into workspace.
__global__ __launch_bounds__(256)
void normalize_rows_f16(const float* __restrict__ x,
                        const float* __restrict__ w,
                        f16* __restrict__ xn,
                        f16* __restrict__ wn) {
  const int row = blockIdx.x;
  const float* src;
  f16* dst;
  if (row < N_ROWS) {
    src = x + (size_t)row * DDIM;
    dst = xn + (size_t)row * DDIM;
  } else {
    const int r = row - N_ROWS;
    src = w + (size_t)r * DDIM;
    dst = wn + (size_t)r * DDIM;
  }
  const int tid = threadIdx.x;                 // 256 threads, 2 floats each
  const float2 v = ((const float2*)src)[tid];
  float s = v.x * v.x + v.y * v.y;
  s = wave_reduce_sum(s);

  __shared__ float part[8];
  const int lane = tid & 31;
  const int wid  = tid >> 5;
  if (lane == 0) part[wid] = s;
  __syncthreads();
  const float tot = part[0] + part[1] + part[2] + part[3] +
                    part[4] + part[5] + part[6] + part[7];
  const float rn = 1.0f / sqrtf(tot);
  dst[2 * tid]     = (f16)(v.x * rn);
  dst[2 * tid + 1] = (f16)(v.y * rn);
}

// 256x128 tile GEMM (f16 WMMA, f32 accum) + fused ArcFace margin epilogue.
// 512 threads = 16 waves; wave grid 8(M) x 2(N); each wave 32x64 = 2x4 WMMA tiles.
// K tiles are double-buffered in LDS and fetched with async global->LDS loads
// (3 x b128 per thread per K-tile: 2 for A, 1 for B).
__global__ __launch_bounds__(512)
void arcface_gemm(const f16* __restrict__ xn,
                  const f16* __restrict__ wn,
                  const int* __restrict__ label,
                  float* __restrict__ out) {
  __shared__ __align__(16) f16 smem[2][TILE_ELEMS];   // [buf][A(256xLDSS) | B(128xLDSS)]

  const int tid   = threadIdx.x;
  const int lane  = tid & 31;
  const int wave  = tid >> 5;
  const int waveM = wave & 7;     // 8 strips of 32 rows
  const int waveN = wave >> 3;    // 2 strips of 64 cols
  const int rowBase = blockIdx.y * BM;
  const int colBase = blockIdx.x * BN;

  v8f acc[2][4];
#pragma unroll
  for (int i = 0; i < 2; ++i)
#pragma unroll
    for (int j = 0; j < 4; ++j) {
      v8f z = {0.f, 0.f, 0.f, 0.f, 0.f, 0.f, 0.f, 0.f};
      acc[i][j] = z;
    }

  // ---- per-thread staging coordinates (3 x 16B chunks per K-tile)
  // A tile: 256 rows x 4 segs = 1024 chunks -> threads tid and tid+512
  // B tile: 128 rows x 4 segs =  512 chunks -> thread tid
  const int seg = (tid & 3) * 8;          // K segment (f16 elems), shared by all 3
  const int rA0 = tid >> 2;               // A rows 0..127
  const int rA1 = rA0 + 128;              // A rows 128..255
  const int rB  = tid >> 2;               // B rows 0..127
  int cB = colBase + rB; if (cB >= NCLS) cB = NCLS - 1;   // clamp edge tile

  const f16* gA0 = xn + (size_t)(rowBase + rA0) * DDIM + seg;
  const f16* gA1 = xn + (size_t)(rowBase + rA1) * DDIM + seg;
  const f16* gB  = wn + (size_t)cB * DDIM + seg;

  uint32_t lA0[2], lA1[2], lB[2];
#pragma unroll
  for (int b = 0; b < 2; ++b) {
    lA0[b] = lds_addr(&smem[b][rA0 * LDSS + seg]);
    lA1[b] = lds_addr(&smem[b][rA1 * LDSS + seg]);
    lB[b]  = lds_addr(&smem[b][BM * LDSS + rB * LDSS + seg]);
  }

  // fragment addressing per CDNA5 16-bit A 16x32 layout:
  // lanes 0-15: rows M=0..15, K 0-7 in v0..3 and K 16-23 in v4..7
  // lanes 16-31: same rows,  K 8-15          and K 24-31
  const int lrow  = lane & 15;
  const int khalf = (lane >> 4) << 3;   // 0 or 8 (f16 elements)

  constexpr int NT = DDIM / BK;         // 16 K-tiles

  // prefetch tile 0 into buffer 0
  async_b128(lA0[0], gA0);
  async_b128(lA1[0], gA1);
  async_b128(lB[0],  gB);

#pragma unroll
  for (int kt = 0; kt < NT; ++kt) {
    wait_async0();        // my async stores to buffer (kt&1) have landed
    __syncthreads();      // everyone's landed; prior tile's ds reads drained

    if (kt + 1 < NT) {    // prefetch next tile into the other buffer
      const int nb = (kt + 1) & 1;
      const size_t off = (size_t)(kt + 1) * BK;   // f16 elements (64B per step)
      async_b128(lA0[nb], gA0 + off);
      async_b128(lA1[nb], gA1 + off);
      async_b128(lB[nb],  gB + off);
    }

    const f16* At = &smem[kt & 1][0];
    const f16* Bt = At + BM * LDSS;

    Frag a[2], b[4];
#pragma unroll
    for (int i = 0; i < 2; ++i) {
      const f16* p = At + (waveM * 32 + i * 16 + lrow) * LDSS + khalf;
      a[i].q[0] = *(const float4*)p;
      a[i].q[1] = *(const float4*)(p + 16);
    }
#pragma unroll
    for (int j = 0; j < 4; ++j) {
      const f16* p = Bt + (waveN * 64 + j * 16 + lrow) * LDSS + khalf;
      b[j].q[0] = *(const float4*)p;
      b[j].q[1] = *(const float4*)(p + 16);
    }

#pragma unroll
    for (int i = 0; i < 2; ++i)
#pragma unroll
      for (int j = 0; j < 4; ++j)
        acc[i][j] = __builtin_amdgcn_wmma_f32_16x16x32_f16(
            false, a[i].v, false, b[j].v, (short)0, acc[i][j], false, false);
  }

  // Epilogue: C/D layout — VGPR v: row = v + (lane<16 ? 0 : 8); col = lane&15.
  const int hi   = (lane >> 4) * 8;
  const int ncol = lane & 15;
#pragma unroll
  for (int i = 0; i < 2; ++i) {
    const int mBase = rowBase + waveM * 32 + i * 16;
    int lbls[8];
#pragma unroll
    for (int v = 0; v < 8; ++v) lbls[v] = label[mBase + hi + v];
#pragma unroll
    for (int j = 0; j < 4; ++j) {
      const int col = colBase + waveN * 64 + j * 16 + ncol;
#pragma unroll
      for (int v = 0; v < 8; ++v) {
        const float c  = acc[i][j][v];
        const float cc = fminf(fmaxf(c, -1.0f + 1e-7f), 1.0f - 1e-7f);
        float val;
        if (col == lbls[v]) {
          if (c > THRESH) {
            const float st = sqrtf(fmaxf(1.0f - cc * cc, 0.0f)); // sin(acos(cc))
            val = SCALE_F * (cc * COSM - st * SINM);             // cos(theta + m)
          } else {
            val = SCALE_F * (cc - MMV);
          }
        } else {
          val = SCALE_F * cc;   // cos(acos(cc)) == cc
        }
        if (col < NCLS)
          out[(size_t)(mBase + hi + v) * NCLS + col] = val;
      }
    }
  }
}

extern "C" void kernel_launch(void* const* d_in, const int* in_sizes, int n_in,
                              void* d_out, int out_size, void* d_ws, size_t ws_size,
                              hipStream_t stream) {
  const float* x     = (const float*)d_in[0];
  const int*   label = (const int*)d_in[1];
  const float* w     = (const float*)d_in[2];
  float*       out   = (float*)d_out;

  f16* xn = (f16*)d_ws;
  f16* wn = xn + (size_t)N_ROWS * DDIM;   // needs ~103 MB workspace total

  normalize_rows_f16<<<N_ROWS + NCLS, 256, 0, stream>>>(x, w, xn, wn);

  dim3 grid((NCLS + BN - 1) / BN, N_ROWS / BM);
  arcface_gemm<<<grid, 512, 0, stream>>>(xn, wn, label, out);
}
